// HofmanSwarm_15015205666877
// MI455X (gfx1250) — matrix operand order repaired and checked by reading, not compiled, not run on hardware
//
#include <hip/hip_runtime.h>
#include <hip/hip_bf16.h>

// Problem constants (match reference)
#define V_  50257
#define VP_ 50304   // V padded to multiple of 64 for 16x64 WMMA super-tiles
#define A_  256
#define D_  512
#define B_  16
#define T_  128
#define NS_ 8

typedef __attribute__((ext_vector_type(16))) __bf16       v16bf;
typedef __attribute__((ext_vector_type(8)))  float        v8f;
typedef __attribute__((ext_vector_type(4)))  unsigned int v4u;

union FragAB {
    v16bf v;
    v4u   q[2];
};

// Load one k-group of fragments (32-bit lane offsets off a uniform base so the
// backend can use SADDR+voffset global loads; k0 folds into the imm offset).
template<int MT, int NT>
__device__ __forceinline__ void gload(FragAB (&a)[MT], FragAB (&b)[NT],
                                      const __bf16* __restrict__ A,
                                      const __bf16* __restrict__ BT,
                                      const unsigned (&aoff)[MT],
                                      const unsigned (&boff)[NT], int k0)
{
#pragma unroll
    for (int i = 0; i < MT; ++i) {
        a[i].q[0] = *reinterpret_cast<const v4u*>(A + aoff[i] + k0);
        a[i].q[1] = *reinterpret_cast<const v4u*>(A + aoff[i] + k0 + 16);
    }
#pragma unroll
    for (int j = 0; j < NT; ++j) {
        b[j].q[0] = *reinterpret_cast<const v4u*>(BT + boff[j] + k0);
        b[j].q[1] = *reinterpret_cast<const v4u*>(BT + boff[j] + k0 + 16);
    }
}

template<int MT, int NT>
__device__ __forceinline__ void gmma(v8f (&acc)[MT][NT], FragAB (&a)[MT], FragAB (&b)[NT])
{
#pragma unroll
    for (int i = 0; i < MT; ++i)
#pragma unroll
        for (int j = 0; j < NT; ++j)
            acc[i][j] = __builtin_amdgcn_wmma_f32_16x16x32_bf16(
                false, a[i].v, false, b[j].v, (short)0, acc[i][j], false, false);
}

// ---------------------------------------------------------------------------
// Register-blocked, double-buffered bf16 WMMA GEMM.
// Each wave computes an (MT*16)x(NT*16) tile:
//   C[m,n] (+)= sum_k A[m,k]*BT[n,k]  (+ bias[n]);  optional bf16 dual-store Cb.
// A : row-major [M,K] bf16 (lda, per-z stride a_zs)
// BT: row-major [Np,K] bf16 = B transposed (ldb, per-z stride b_zs)
// C : fp32 (ldc, per-z stride c_zs).  K multiple of 64.
// Fragment layouts per CDNA5 ISA 7.12.2 (A 16x32, B 32x16, C/D f32 16x16).
// ---------------------------------------------------------------------------
template<int MT, int NT>
__global__ __launch_bounds__(128)
void wmma_gemm_bf16(const __bf16* __restrict__ A, int lda, long long a_zs,
                    const __bf16* __restrict__ BT, int ldb, long long b_zs,
                    float* __restrict__ C, long long ldc, long long c_zs,
                    __bf16* __restrict__ Cb,            // optional bf16 copy of C
                    const float* __restrict__ bias,
                    int N, int K, int n_limit, int accumulate)
{
    const int lane = threadIdx.x;                         // 0..31, one wave
    const int nsup = blockIdx.x * blockDim.y + threadIdx.y;
    const int n0   = nsup * (NT * 16);
    if (n0 >= N) return;                                  // wave-uniform exit
    const int m0   = blockIdx.y * (MT * 16);

    A  += (long long)blockIdx.z * a_zs;
    BT += (long long)blockIdx.z * b_zs;
    C  += (long long)blockIdx.z * c_zs;
    if (Cb) Cb += (long long)blockIdx.z * c_zs;

    const int sub  = lane & 15;
    const int half = lane >> 4;
    const int mb   = half * 8;

    v8f acc[MT][NT];
#pragma unroll
    for (int i = 0; i < MT; ++i)
#pragma unroll
        for (int j = 0; j < NT; ++j)
#pragma unroll
            for (int r = 0; r < 8; ++r) acc[i][j][r] = 0.f;

    if (accumulate) {
#pragma unroll
        for (int i = 0; i < MT; ++i)
#pragma unroll
            for (int j = 0; j < NT; ++j) {
                const int n = n0 + j * 16 + sub;
                if (n < n_limit) {
#pragma unroll
                    for (int r = 0; r < 8; ++r)
                        acc[i][j][r] = C[(long long)(m0 + i * 16 + mb + r) * ldc + n];
                }
            }
    }

    // 32-bit lane offsets (all operands < 2^31 elements)
    unsigned aoff[MT], boff[NT];
#pragma unroll
    for (int i = 0; i < MT; ++i)
        aoff[i] = (unsigned)((m0 + i * 16 + sub) * lda + half * 8);
#pragma unroll
    for (int j = 0; j < NT; ++j)
        boff[j] = (unsigned)((n0 + j * 16 + sub) * ldb + half * 8);

    // Double-buffered main loop (K % 64 == 0): loads of group k+32 overlap
    // the WMMAs of group k, so waits never drain to loadcnt==0 mid-loop.
    FragAB a0[MT], b0[NT], a1[MT], b1[NT];
    gload<MT, NT>(a0, b0, A, BT, aoff, boff, 0);
    for (int k0 = 0; k0 < K; k0 += 64) {
        gload<MT, NT>(a1, b1, A, BT, aoff, boff, k0 + 32);
        gmma<MT, NT>(acc, a0, b0);
        if (k0 + 64 < K) gload<MT, NT>(a0, b0, A, BT, aoff, boff, k0 + 64);
        gmma<MT, NT>(acc, a1, b1);
    }

#pragma unroll
    for (int j = 0; j < NT; ++j) {
        const int n = n0 + j * 16 + sub;
        if (n < n_limit) {
            const float bv = bias ? bias[n] : 0.f;
#pragma unroll
            for (int i = 0; i < MT; ++i)
#pragma unroll
                for (int r = 0; r < 8; ++r) {
                    const long long ci = (long long)(m0 + i * 16 + mb + r) * ldc + n;
                    const float v = acc[i][j][r] + bv;
                    C[ci] = v;
                    if (Cb) Cb[ci] = (__bf16)v;
                }
        }
    }
}

// ---------------------------------------------------------------------------
// Setup kernels (run once per launch)
// ---------------------------------------------------------------------------
__global__ void softmax_iw_kernel(const float* __restrict__ imp, float* __restrict__ iw)
{
    __shared__ float red[256];
    const int t = threadIdx.x;                 // A_ == 256 threads
    const float v = imp[t];
    red[t] = v; __syncthreads();
    for (int s = 128; s > 0; s >>= 1) { if (t < s) red[t] = fmaxf(red[t], red[t + s]); __syncthreads(); }
    const float m = red[0]; __syncthreads();
    const float e = expf(v - m);
    red[t] = e; __syncthreads();
    for (int s = 128; s > 0; s >>= 1) { if (t < s) red[t] += red[t + s]; __syncthreads(); }
    iw[t] = e / red[0];
}

__global__ void conv_bf16_kernel(const float* __restrict__ src, __bf16* __restrict__ dst, long n)
{
    const long i = blockIdx.x * (long)blockDim.x + threadIdx.x;
    if (i < n) dst[i] = (__bf16)src[i];
}

// src: fp32 row-major [K, N]; dst: bf16 row-major [Np, K]; zero-fills n >= N.
__global__ void transpose_to_bf16(const float* __restrict__ src, __bf16* __restrict__ dst,
                                  int K, int N, int Np)
{
    const long i = blockIdx.x * (long)blockDim.x + threadIdx.x;
    if (i >= (long)Np * K) return;
    const int k = (int)(i % K);
    const int n = (int)(i / K);
    dst[i] = (n < N) ? (__bf16)src[(long)k * N + n] : (__bf16)0.f;
}

// ---------------------------------------------------------------------------
// Per-step kernels
// ---------------------------------------------------------------------------
__global__ void compute_pt_kernel(const int* __restrict__ idx,
                                  const float* __restrict__ perception,
                                  const float* __restrict__ pos_emb,
                                  int t, __bf16* __restrict__ pt)
{
    const int i = blockIdx.x * blockDim.x + threadIdx.x;    // B_*D_
    if (i >= B_ * D_) return;
    const int b = i / D_, d = i % D_;
    const int tok = idx[b * T_ + t];
    pt[i] = (__bf16)(perception[(long)tok * D_ + d] + pos_emb[(long)t * D_ + d]);
}

// Refresh bf16 transposed copy of the NS sensory rows after the sensory add.
// (Row-major bf16 copy is produced by the sensory GEMM's dual-store.)
__global__ void conv_sensory_kernel(const float* __restrict__ st,
                                    __bf16* __restrict__ stT)
{
    const int i = blockIdx.x * blockDim.x + threadIdx.x;    // B_*NS_*D_
    if (i >= B_ * NS_ * D_) return;
    const int d   = i % D_;
    const int rem = i / D_;
    const int a   = rem % NS_;
    const int b   = rem / NS_;
    stT[((long)b * D_ + d) * A_ + a] = (__bf16)st[((long)b * A_ + a) * D_ + d];
}

// gelu(candraw+bc), sigmoid(gateraw+bg), gate-mix with states, LayerNorm(D)
// -> states fp32 + bf16 row-major + bf16 per-batch transposed
__global__ void combine_ln_kernel(const float* __restrict__ candraw,
                                  const float* __restrict__ gateraw,
                                  float* __restrict__ states,
                                  __bf16* __restrict__ statesB,
                                  __bf16* __restrict__ statesT,
                                  const float* __restrict__ bc,
                                  const float* __restrict__ bg,
                                  const float* __restrict__ gamma,
                                  const float* __restrict__ beta)
{
    const int  r    = blockIdx.x;              // row in [0, B*A)
    const int  tid  = threadIdx.x;             // 256 threads, D=512 -> 2 per thread
    const int  b    = r / A_;
    const int  a    = r % A_;
    const long base = (long)r * D_;
    __shared__ float red[256];

    float h[2];
    float lsum = 0.f;
#pragma unroll
    for (int j = 0; j < 2; ++j) {
        const int d = tid + j * 256;
        const float craw = candraw[base + d] + bc[d];
        const float cand = 0.5f * craw * (1.f + erff(craw * 0.70710678118654752f));
        const float graw = gateraw[base + d] + bg[d];
        const float gate = 1.f / (1.f + expf(-graw));
        const float s    = states[base + d];
        h[j] = gate * cand + (1.f - gate) * s;
        lsum += h[j];
    }
    red[tid] = lsum; __syncthreads();
    for (int sreduce = 128; sreduce > 0; sreduce >>= 1) {
        if (tid < sreduce) red[tid] += red[tid + sreduce];
        __syncthreads();
    }
    const float mu = red[0] / (float)D_;
    __syncthreads();
    float lvar = 0.f;
#pragma unroll
    for (int j = 0; j < 2; ++j) { const float dd = h[j] - mu; lvar += dd * dd; }
    red[tid] = lvar; __syncthreads();
    for (int sreduce = 128; sreduce > 0; sreduce >>= 1) {
        if (tid < sreduce) red[tid] += red[tid + sreduce];
        __syncthreads();
    }
    const float inv = rsqrtf(red[0] / (float)D_ + 1e-5f);
#pragma unroll
    for (int j = 0; j < 2; ++j) {
        const int d = tid + j * 256;
        const float o = (h[j] - mu) * inv * gamma[d] + beta[d];
        states[base + d]  = o;
        statesB[base + d] = (__bf16)o;
        statesT[((long)b * D_ + d) * A_ + a] = (__bf16)o;
    }
}

// cons[b,d] = sum_a iw[a] * states[b,a,d]
__global__ void cons_kernel(const float* __restrict__ states,
                            const float* __restrict__ iw,
                            float* __restrict__ cons)
{
    const int i = blockIdx.x * blockDim.x + threadIdx.x;   // B_*D_
    if (i >= B_ * D_) return;
    const int b = i / D_, d = i % D_;
    const float* sp = states + (long)b * A_ * D_ + d;
    float acc = 0.f;
    for (int a = 0; a < A_; ++a) acc += iw[a] * sp[(long)a * D_];
    cons[i] = acc;
}

// hcons = gelu(cons @ W1 + b1) -> bf16  (16 x 1024, tiny)
__global__ void mlp1_kernel(const float* __restrict__ cons,
                            const float* __restrict__ W1,
                            const float* __restrict__ b1,
                            __bf16* __restrict__ hcons)
{
    const int i = blockIdx.x * blockDim.x + threadIdx.x;   // B_*2D_
    if (i >= B_ * 2 * D_) return;
    const int b = i / (2 * D_), n = i % (2 * D_);
    const float* cp = cons + (long)b * D_;
    float acc = b1[n];
    for (int k = 0; k < D_; ++k) acc += cp[k] * W1[(long)k * 2 * D_ + n];
    hcons[i] = (__bf16)(0.5f * acc * (1.f + erff(acc * 0.70710678118654752f)));
}

// ---------------------------------------------------------------------------
// Host driver
// ---------------------------------------------------------------------------
static inline size_t align256(size_t x) { return (x + 255) & ~(size_t)255; }

extern "C" void kernel_launch(void* const* d_in, const int* in_sizes, int n_in,
                              void* d_out, int out_size, void* d_ws, size_t ws_size,
                              hipStream_t stream)
{
    (void)in_sizes; (void)n_in; (void)out_size; (void)ws_size;

    const int*   idx        = (const int*)  d_in[0];
    const float* perception = (const float*)d_in[1];
    const float* pos_emb    = (const float*)d_in[2];
    const float* Ws         = (const float*)d_in[3];
    const float* bs         = (const float*)d_in[4];
    const float* Wnet       = (const float*)d_in[5];
    const float* Wg         = (const float*)d_in[6];
    const float* bg         = (const float*)d_in[7];
    const float* Wc         = (const float*)d_in[8];
    const float* bc         = (const float*)d_in[9];
    const float* gamma      = (const float*)d_in[10];
    const float* beta       = (const float*)d_in[11];
    const float* imp        = (const float*)d_in[12];
    const float* W1         = (const float*)d_in[13];
    const float* b1         = (const float*)d_in[14];
    const float* W2         = (const float*)d_in[15];
    const float* b2         = (const float*)d_in[16];
    float* out = (float*)d_out;

    // Workspace carve-up (~149 MB total)
    char* p = (char*)d_ws;
    auto carve = [&](size_t bytes) { char* r = p; p += align256(bytes); return r; };
    float*  iw      = (float*) carve((size_t)A_ * 4);
    __bf16* pt      = (__bf16*)carve((size_t)B_ * D_ * 2);
    __bf16* WnetB   = (__bf16*)carve((size_t)A_ * A_ * 2);
    __bf16* WsT     = (__bf16*)carve((size_t)NS_ * D_ * D_ * 2);     // [NS*D, D]
    __bf16* WcT     = (__bf16*)carve((size_t)D_ * D_ * 2);           // [D, D]
    __bf16* WgT     = (__bf16*)carve((size_t)D_ * 2 * D_ * 2);       // [D, 2D]
    __bf16* W2T     = (__bf16*)carve((size_t)VP_ * 2 * D_ * 2);      // [VP, 2D], L2-resident (~98 MB)
    float*  states  = (float*) carve((size_t)B_ * A_ * D_ * 4);
    __bf16* statesB = (__bf16*)carve((size_t)B_ * A_ * D_ * 2);      // row-major [B*A, D]
    __bf16* statesT = (__bf16*)carve((size_t)B_ * D_ * A_ * 2);      // per-batch [D, A]
    float*  inter   = (float*) carve((size_t)B_ * A_ * D_ * 4);
    __bf16* interB  = (__bf16*)carve((size_t)B_ * A_ * D_ * 2);
    float*  candraw = (float*) carve((size_t)B_ * A_ * D_ * 4);
    float*  gateraw = (float*) carve((size_t)B_ * A_ * D_ * 4);
    float*  cons    = (float*) carve((size_t)B_ * D_ * 4);
    __bf16* hcons   = (__bf16*)carve((size_t)B_ * 2 * D_ * 2);

    // ---- one-time setup ----
    softmax_iw_kernel<<<1, 256, 0, stream>>>(imp, iw);
    { const long n = (long)A_ * A_;
      conv_bf16_kernel<<<(unsigned)((n + 255) / 256), 256, 0, stream>>>(Wnet, WnetB, n); }
    transpose_to_bf16<<<(unsigned)(((long)NS_ * D_ * D_ + 255) / 256), 256, 0, stream>>>(Ws, WsT, D_, NS_ * D_, NS_ * D_);
    transpose_to_bf16<<<(unsigned)(((long)D_ * D_ + 255) / 256), 256, 0, stream>>>(Wc, WcT, D_, D_, D_);
    transpose_to_bf16<<<(unsigned)(((long)D_ * 2 * D_ + 255) / 256), 256, 0, stream>>>(Wg, WgT, 2 * D_, D_, D_);
    transpose_to_bf16<<<(unsigned)(((long)VP_ * 2 * D_ + 255) / 256), 256, 0, stream>>>(W2, W2T, 2 * D_, V_, VP_);
    hipMemsetAsync(states,  0, (size_t)B_ * A_ * D_ * 4, stream);
    hipMemsetAsync(statesB, 0, (size_t)B_ * A_ * D_ * 2, stream);
    hipMemsetAsync(statesT, 0, (size_t)B_ * D_ * A_ * 2, stream);

    const dim3 blk(32, 4);   // 4 waves/block, one super-tile per wave

    for (int t = 0; t < T_; ++t) {
        // p_t (bf16)
        compute_pt_kernel<<<(B_ * D_ + 255) / 256, 256, 0, stream>>>(idx, perception, pos_emb, t, pt);

        // states[:, :NS, :] += p_t @ Ws + bs   (M=16, N=4096, K=512; 16x64 tiles)
        // dual-store refreshes the bf16 row-major copy of the touched rows
        wmma_gemm_bf16<1, 4><<<dim3(16, 1, 1), blk, 0, stream>>>(
            pt, D_, 0, WsT, D_, 0,
            states, (long long)A_ * D_, 0, statesB,
            bs, NS_ * D_, D_, NS_ * D_, 1);

        // refresh bf16 transposed copy of the NS sensory rows
        conv_sensory_kernel<<<(B_ * NS_ * D_ + 255) / 256, 256, 0, stream>>>(states, statesT);

        // inter[b] = W_net @ states[b]   (M=256, N=512, K=256, z=B; 32x32 tiles)
        // dual-store produces interB (bf16) in the epilogue
        wmma_gemm_bf16<2, 2><<<dim3(4, 8, B_), blk, 0, stream>>>(
            WnetB, A_, 0, statesT, A_, (long long)D_ * A_,
            inter, D_, (long long)A_ * D_, interB,
            nullptr, D_, A_, D_, 0);

        // candraw = inter @ Wc            (M=4096, N=512, K=512; 32x32 tiles)
        wmma_gemm_bf16<2, 2><<<dim3(4, 128, 1), blk, 0, stream>>>(
            interB, D_, 0, WcT, D_, 0, candraw, D_, 0, nullptr,
            nullptr, D_, D_, D_, 0);

        // gateraw = states @ Wg[:D,:] + inter @ Wg[D:,:]   (split-K via accumulate)
        wmma_gemm_bf16<2, 2><<<dim3(4, 128, 1), blk, 0, stream>>>(
            statesB, D_, 0, WgT, 2 * D_, 0, gateraw, D_, 0, nullptr,
            nullptr, D_, D_, D_, 0);
        wmma_gemm_bf16<2, 2><<<dim3(4, 128, 1), blk, 0, stream>>>(
            interB, D_, 0, WgT + D_, 2 * D_, 0, gateraw, D_, 0, nullptr,
            nullptr, D_, D_, D_, 1);

        // gelu/sigmoid/mix/LayerNorm -> states (fp32 + both bf16 copies)
        combine_ln_kernel<<<B_ * A_, 256, 0, stream>>>(
            candraw, gateraw, states, statesB, statesT, bc, bg, gamma, beta);

        // consensus + tiny MLP1
        cons_kernel<<<(B_ * D_ + 255) / 256, 256, 0, stream>>>(states, iw, cons);
        mlp1_kernel<<<(B_ * 2 * D_ + 255) / 256, 256, 0, stream>>>(cons, W1, b1, hcons);

        // logits[:, t, :] = hcons @ W2 + b2   (M=16, N=VP=50304, K=1024; 16x64 tiles)
        wmma_gemm_bf16<1, 4><<<dim3((VP_ / 64 + 3) / 4, 1, 1), blk, 0, stream>>>(
            hcons, 2 * D_, 0, W2T, 2 * D_, 0,
            out + (long long)t * V_, (long long)T_ * V_, 0, nullptr,
            b2, VP_, 2 * D_, V_, 0);
    }
}
